// ModelWithgraph_embedding_modif_gcnconv_61022895342233
// MI455X (gfx1250) — matrix operand order repaired and checked by reading, not compiled, not run on hardware
//
#include <hip/hip_runtime.h>
#include <hip/hip_bf16.h>

// ---------------- problem constants (from reference) ----------------
#define NN 50000
#define EE 600000
#define DD 128
#define HM 256
#define CC 32
#define GG 512
#define LL 3

typedef __bf16 bf16_t;
typedef __bf16 v16bf __attribute__((ext_vector_type(16)));
typedef __bf16 v8bf  __attribute__((ext_vector_type(8)));
typedef float  v8f   __attribute__((ext_vector_type(8)));

__device__ __forceinline__ bf16_t f2bf(float f) {
    unsigned u = __builtin_bit_cast(unsigned, f);
    u += 0x7FFFu + ((u >> 16) & 1u);           // round-to-nearest-even
    unsigned short h = (unsigned short)(u >> 16);
    return __builtin_bit_cast(bf16_t, h);
}

// ---------------- utility kernels ----------------
__global__ void zero_f32(float* p, long n) {
    long i = (long)blockIdx.x * blockDim.x + threadIdx.x;
    if (i < n) p[i] = 0.0f;
}

__global__ void cvt_to_bf16(const float* __restrict__ x, bf16_t* __restrict__ xb, long n) {
    long i = (long)blockIdx.x * blockDim.x + threadIdx.x;
    if (i < n) xb[i] = f2bf(x[i]);
}

// ---------------- weight repack into WMMA B-fragment order ----------------
// dst[(t*kcn + kc)*512 + lane*16 + e] ; lane 0-15: col=t*16+lane, K=kc*32+e
//                                      lane 16-31: col=t*16+(lane-16), K=kc*32+16+e
__global__ void pack_fused(const float* __restrict__ Ws, const float* __restrict__ Wd,
                           const float* __restrict__ Wt, const float* __restrict__ Wi,
                           bf16_t* __restrict__ dst) {
    int idx = blockIdx.x * 256 + threadIdx.x;
    if (idx >= LL * 32 * 4 * 512) return;
    int e    = idx & 15;
    int lane = (idx >> 4) & 31;
    int kc   = (idx >> 9) & 3;
    int t    = (idx >> 11) & 31;
    int l    = idx >> 16;
    int n = t * 16 + (lane & 15);
    int k = kc * 32 + (lane >> 4) * 16 + e;
    int sel = n >> 7, c = n & 127;
    const float* W = (sel == 0) ? Ws : (sel == 1) ? Wd : (sel == 2) ? Wt : Wi;
    dst[idx] = f2bf(W[l * (DD * DD) + k * DD + c]);
}

__global__ void pack_gen(const float* __restrict__ W, int Ncols,
                         bf16_t* __restrict__ dst, int ntiles, int kcn) {
    int idx = blockIdx.x * 256 + threadIdx.x;
    if (idx >= ntiles * kcn * 512) return;
    int e    = idx & 15;
    int lane = (idx >> 4) & 31;
    int frag = idx >> 9;
    int kc = frag % kcn;
    int t  = frag / kcn;
    int n = t * 16 + (lane & 15);
    int k = kc * 32 + (lane >> 4) * 16 + e;
    dst[idx] = f2bf(W[k * Ncols + n]);
}

// ---------------- bf16 WMMA GEMM: one wave per 16x(4x16) tile strip ----------
// A: row-major [M x K] bf16 (K multiple of 32), Bp: packed fragments.
// Each wave keeps 4 accumulators (4 consecutive n-tiles) and reuses each A
// fragment across 4 WMMAs.
__global__ void gemm_wmma_bf16(const bf16_t* __restrict__ A, int K, int kcn,
                               const bf16_t* __restrict__ Bp, int ntiles,
                               const float* __restrict__ bias,
                               float* __restrict__ Cf, bf16_t* __restrict__ Cb,
                               int ldc, int dorelu) {
    int wave = threadIdx.x >> 5;
    int lane = threadIdx.x & 31;
    int tg = blockIdx.y * 4 + wave;        // tile-group of 4 n-tiles
    int nbase = tg * 4;
    if (nbase >= ntiles) return;           // uniform per wave
    int half = lane >> 4, r = lane & 15;
    int m0 = blockIdx.x * 16;

    const bf16_t* Arow = A + (size_t)(m0 + r) * K + half * 16;
    const bf16_t* Bf[4];
#pragma unroll
    for (int j = 0; j < 4; ++j) {
        int t = nbase + j;
        if (t > ntiles - 1) t = ntiles - 1;          // clamp: dummy in-bounds reads
        Bf[j] = Bp + (size_t)t * kcn * 512 + lane * 16;
    }

    v8f acc[4];
#pragma unroll
    for (int j = 0; j < 4; ++j) acc[j] = (v8f){0.f, 0.f, 0.f, 0.f, 0.f, 0.f, 0.f, 0.f};

    for (int kc = 0; kc < kcn; ++kc) {
        v8bf alo = *(const v8bf*)(Arow + kc * 32);        // K = kc*32 + half*16 + 0..7
        v8bf ahi = *(const v8bf*)(Arow + kc * 32 + 8);    // K = kc*32 + half*16 + 8..15
        v16bf a;
#pragma unroll
        for (int i = 0; i < 8; ++i) { a[i] = alo[i]; a[i + 8] = ahi[i]; }
#pragma unroll
        for (int j = 0; j < 4; ++j) {
            v16bf b = *(const v16bf*)(Bf[j] + kc * 512);
            acc[j] = __builtin_amdgcn_wmma_f32_16x16x32_bf16(false, a, false, b,
                                                             (short)0, acc[j], false, false);
        }
    }

#pragma unroll
    for (int j = 0; j < 4; ++j) {
        int ntile = nbase + j;
        if (ntile >= ntiles) break;
        int n0 = ntile * 16;
        float bv = bias ? bias[n0 + r] : 0.0f;
#pragma unroll
        for (int g = 0; g < 8; ++g) {
            int row = m0 + g + 8 * half;              // C layout: VGPR g, lane half
            float v = acc[j][g] + bv;
            if (dorelu) v = fmaxf(v, 0.0f);
            size_t o = (size_t)row * ldc + n0 + r;
            if (Cf) Cf[o] = v;
            if (Cb) Cb[o] = f2bf(v);
        }
    }
}

// ---------------- graph degree / normalization ----------------
__global__ void deg_kernel(const int* __restrict__ ei, const int* __restrict__ ea,
                           float* __restrict__ deg) {
    int e = blockIdx.x * blockDim.x + threadIdx.x;
    if (e >= EE) return;
    int col = ei[EE + e];
#pragma unroll
    for (int r = 0; r < 3; ++r)
        if (ea[e * 3 + r] == 1) unsafeAtomicAdd(&deg[r * NN + col], 1.0f);
}

__global__ void dinv_kernel(float* deg) {
    int i = blockIdx.x * blockDim.x + threadIdx.x;
    if (i >= 3 * NN) return;
    float d = deg[i];
    deg[i] = (d > 0.0f) ? rsqrtf(d) : 0.0f;
}

// ---------------- edge scatter: one wave per (edge, relation) ----------------
__global__ void scatter_edges(const int* __restrict__ ei, const int* __restrict__ ea,
                              const float* __restrict__ dinv, const float* __restrict__ H,
                              float* __restrict__ agg) {
    long wid = ((long)blockIdx.x * blockDim.x + threadIdx.x) >> 5;
    int lane = threadIdx.x & 31;
    if (wid >= (long)3 * EE) return;
    int r = (int)(wid / EE);
    int e = (int)(wid - (long)r * EE);
    if (ea[e * 3 + r] != 1) return;
    int row = ei[e], col = ei[EE + e];
    float nm = dinv[r * NN + row] * dinv[r * NN + col];
    const float* hs = H + (size_t)row * 512 + r * 128;
    float* as = agg + (size_t)col * 384 + r * 128;
#pragma unroll
    for (int i = 0; i < 4; ++i)
        unsafeAtomicAdd(as + lane + i * 32, nm * hs[lane + i * 32]);
}

// ---------------- combine relations + BN pass 1 ----------------
__global__ void combine_bn1(const float* __restrict__ H, const float* __restrict__ agg,
                            const float* __restrict__ bs, const float* __restrict__ bd,
                            const float* __restrict__ bt, const float* __restrict__ bi,
                            float* __restrict__ xtmp, float* __restrict__ bnsum,
                            float* __restrict__ bnsq) {
    int d = threadIdx.x;                 // 128 features
    int n0 = blockIdx.x * 64;
    float vbs = bs[d], vbd = bd[d], vbt = bt[d], vbi = bi[d];
    float s1 = 0.f, s2 = 0.f;
    for (int i = 0; i < 64; ++i) {
        int n = n0 + i;
        if (n >= NN) break;
        const float* a = agg + (size_t)n * 384;
        float y = fmaxf(a[d] + vbs, 0.f) + fmaxf(a[128 + d] + vbd, 0.f) +
                  fmaxf(a[256 + d] + vbt, 0.f) +
                  fmaxf(H[(size_t)n * 512 + 384 + d] + vbi, 0.f);
        xtmp[(size_t)n * 128 + d] = y;
        s1 += y; s2 += y * y;
    }
    unsafeAtomicAdd(&bnsum[d], s1);
    unsafeAtomicAdd(&bnsq[d], s2);
}

__global__ void bn_apply(const float* __restrict__ xtmp, const float* __restrict__ bnsum,
                         const float* __restrict__ bnsq, const float* __restrict__ gamma,
                         const float* __restrict__ beta, float* __restrict__ xcur,
                         bf16_t* __restrict__ xb) {
    long idx = (long)blockIdx.x * blockDim.x + threadIdx.x;
    if (idx >= (long)NN * 128) return;
    int d = (int)(idx & 127);
    float inv = 1.0f / (float)NN;
    float mu = bnsum[d] * inv;
    float var = bnsq[d] * inv - mu * mu;
    float y = (xtmp[idx] - mu) * rsqrtf(var + 1e-5f) * gamma[d] + beta[d];
    y = fmaxf(y, 0.f);
    xcur[idx] = y;
    xb[idx] = f2bf(y);
}

// ---------------- pooling + concat ----------------
__global__ void pool_add(const float* __restrict__ xcur, const int* __restrict__ batch,
                         float* __restrict__ pooled) {
    int d = threadIdx.x;
    int n = blockIdx.x;
    unsafeAtomicAdd(&pooled[(size_t)batch[n] * 128 + d], xcur[(size_t)n * 128 + d]);
}

__global__ void concat_bf16(const float* __restrict__ xcur, const float* __restrict__ pooled,
                            const int* __restrict__ batch, bf16_t* __restrict__ xcb) {
    long idx = (long)blockIdx.x * blockDim.x + threadIdx.x;
    if (idx >= (long)NN * 256) return;
    int n = (int)(idx >> 8);
    int d = (int)(idx & 255);
    float v = (d < 128) ? xcur[(size_t)n * 128 + d]
                        : pooled[(size_t)batch[n] * 128 + (d - 128)];
    xcb[idx] = f2bf(v);
}

// ---------------- host orchestration ----------------
extern "C" void kernel_launch(void* const* d_in, const int* in_sizes, int n_in,
                              void* d_out, int out_size, void* d_ws, size_t ws_size,
                              hipStream_t stream) {
    const float* x      = (const float*)d_in[0];
    const int*   ei     = (const int*)d_in[1];
    const int*   ea     = (const int*)d_in[2];
    const int*   batch  = (const int*)d_in[3];
    const float* W_s    = (const float*)d_in[4];
    const float* b_s    = (const float*)d_in[5];
    const float* W_d    = (const float*)d_in[6];
    const float* b_d    = (const float*)d_in[7];
    const float* W_t    = (const float*)d_in[8];
    const float* b_t    = (const float*)d_in[9];
    const float* W_i    = (const float*)d_in[10];
    const float* b_i    = (const float*)d_in[11];
    const float* gamma  = (const float*)d_in[12];
    const float* beta   = (const float*)d_in[13];
    const float* W_fc1  = (const float*)d_in[14];
    const float* b_fc1  = (const float*)d_in[15];
    const float* W_fc2  = (const float*)d_in[16];
    const float* b_fc2  = (const float*)d_in[17];

    char* ws = (char*)d_ws;
    size_t off = 0;
    auto alloc = [&](size_t bytes) -> void* {
        off = (off + 255) & ~(size_t)255;
        void* p = ws + off;
        off += bytes;
        return p;
    };
    bf16_t* WpackL = (bf16_t*)alloc((size_t)LL * 32 * 4 * 512 * 2);
    bf16_t* Wpack1 = (bf16_t*)alloc((size_t)16 * 8 * 512 * 2);
    bf16_t* Wpack2 = (bf16_t*)alloc((size_t)2 * 8 * 512 * 2);
    bf16_t* xb     = (bf16_t*)alloc((size_t)NN * 128 * 2);
    float*  H      = (float*)alloc((size_t)NN * 512 * 4);
    float*  agg    = (float*)alloc((size_t)NN * 384 * 4);
    float*  xtmp   = (float*)alloc((size_t)NN * 128 * 4);
    float*  xcur   = (float*)alloc((size_t)NN * 128 * 4);
    float*  deg    = (float*)alloc((size_t)3 * NN * 4);
    float*  bnsum  = (float*)alloc(128 * 4);
    float*  bnsq   = (float*)alloc(128 * 4);
    float*  pooled = (float*)alloc((size_t)GG * 128 * 4);
    // aliases (safe: H/agg dead after last combine_bn1)
    bf16_t* xcb = (bf16_t*)H;     // [N,256] bf16 = 25.6MB <= 102.4MB
    bf16_t* h1b = (bf16_t*)agg;   // [N,256] bf16 = 25.6MB <= 76.8MB

    // weight repack (tiny)
    pack_fused<<<768, 256, 0, stream>>>(W_s, W_d, W_t, W_i, WpackL);
    pack_gen<<<256, 256, 0, stream>>>(W_fc1, HM, Wpack1, 16, 8);
    pack_gen<<<32, 256, 0, stream>>>(W_fc2, CC, Wpack2, 2, 8);

    // initial activation -> bf16
    cvt_to_bf16<<<25000, 256, 0, stream>>>(x, xb, (long)NN * 128);

    // degrees / normalization (identical across layers)
    zero_f32<<<(3 * NN + 255) / 256, 256, 0, stream>>>(deg, (long)3 * NN);
    deg_kernel<<<(EE + 255) / 256, 256, 0, stream>>>(ei, ea, deg);
    dinv_kernel<<<(3 * NN + 255) / 256, 256, 0, stream>>>(deg);

    const int MT = NN / 16; // 3125, exact
    for (int l = 0; l < LL; ++l) {
        // fused X @ [Ws|Wd|Wt|Wi] -> H [N,512]  (32 n-tiles = 8 tile-groups)
        gemm_wmma_bf16<<<dim3(MT, 2), 128, 0, stream>>>(
            xb, DD, 4, WpackL + (size_t)l * 32 * 4 * 512, 32,
            nullptr, H, nullptr, 512, 0);
        // edge aggregation
        zero_f32<<<(int)(((long)NN * 384 + 255) / 256), 256, 0, stream>>>(agg, (long)NN * 384);
        {
            long waves = (long)3 * EE;
            int blocks = (int)((waves * 32 + 255) / 256);
            scatter_edges<<<blocks, 256, 0, stream>>>(ei, ea, deg, H, agg);
        }
        // combine + BN
        zero_f32<<<1, 256, 0, stream>>>(bnsum, 128);
        zero_f32<<<1, 256, 0, stream>>>(bnsq, 128);
        combine_bn1<<<(NN + 63) / 64, 128, 0, stream>>>(
            H, agg, b_s + l * DD, b_d + l * DD, b_t + l * DD, b_i + l * DD,
            xtmp, bnsum, bnsq);
        bn_apply<<<25000, 256, 0, stream>>>(xtmp, bnsum, bnsq,
                                            gamma + l * DD, beta + l * DD, xcur, xb);
    }

    // global_add_pool + concat
    zero_f32<<<(GG * 128 + 255) / 256, 256, 0, stream>>>(pooled, (long)GG * 128);
    pool_add<<<NN, 128, 0, stream>>>(xcur, batch, pooled);
    concat_bf16<<<50000, 256, 0, stream>>>(xcur, pooled, batch, xcb);

    // fc1: [N,256]@[256,256] + bias, relu -> bf16  (16 n-tiles = 4 groups)
    gemm_wmma_bf16<<<dim3(MT, 1), 128, 0, stream>>>(
        xcb, 256, 8, Wpack1, 16, b_fc1, nullptr, h1b, 256, 1);
    // fc2: [N,256]@[256,32] + bias -> d_out (fp32)  (2 n-tiles = 1 group)
    gemm_wmma_bf16<<<dim3(MT, 1), 128, 0, stream>>>(
        h1b, 256, 8, Wpack2, 2, b_fc2, (float*)d_out, nullptr, 32, 0);
}
